// AffineCoupling_68874095559357
// MI455X (gfx1250) — matrix operand order repaired and checked by reading, not compiled, not run on hardware
//
#include <hip/hip_runtime.h>
#include <hip/hip_bf16.h>
#include <stdint.h>

// ---------------------------------------------------------------------------
// Affine coupling (RealNVP) fused kernel for MI455X / gfx1250.
// s,t MLPs computed with v_wmma_f32_16x16x32_bf16, fp32 accumulation.
// ---------------------------------------------------------------------------

typedef __attribute__((ext_vector_type(16))) __bf16 v16bf;
typedef __attribute__((ext_vector_type(8)))  float  v8f;

#define BROWS 16384
#define DCOL  1024
#define HALFD 512
#define HIDD  4096

#define MT       32            // rows of z per workgroup
#define KC       128           // HID chunk processed per fused iteration
#define LDA      (HALFD + 8)   // 520 bf16; 4*m bank rotation, conflict-min
#define LDH      (KC + 8)      // 136 bf16
#define NTHREADS 256           // 8 waves (wave32)

static __device__ __forceinline__ v16bf load_frag(const void* p0, const void* p1) {
  union { v16bf v; uint4 u[2]; } f;
  f.u[0] = *(const uint4*)p0;   // K block 0 (8 bf16, 16B)
  f.u[1] = *(const uint4*)p1;   // K block 1 (8 bf16, 16B)
  return f.v;
}

static __device__ __forceinline__ v8f vzero8() {
  v8f v = {0.f, 0.f, 0.f, 0.f, 0.f, 0.f, 0.f, 0.f};
  return v;
}

// ---------------------------------------------------------------------------
// Pre-pass: fp32 (R x C) row-major  ->  bf16 (C x R) row-major (transposed).
// Transposed storage makes WMMA B-operand loads contiguous per lane.
// ---------------------------------------------------------------------------
__global__ __launch_bounds__(256)
void transpose_cvt_kernel(const float* __restrict__ in, __bf16* __restrict__ out,
                          int R, int C) {
  __shared__ float tile[32][33];
  const int tx = threadIdx.x;          // 0..31
  const int ty = threadIdx.y;          // 0..7
  const int bc = blockIdx.x * 32;      // col base in 'in'
  const int br = blockIdx.y * 32;      // row base in 'in'
  #pragma unroll
  for (int j = ty; j < 32; j += 8)
    tile[j][tx] = in[(size_t)(br + j) * C + (bc + tx)];
  __syncthreads();
  #pragma unroll
  for (int j = ty; j < 32; j += 8)
    out[(size_t)(bc + j) * R + (br + tx)] = (__bf16)tile[tx][j];
}

// ---------------------------------------------------------------------------
// One MLP branch: acc(16x128 per wave, fp32) = (relu(A @ W1 + b1)) @ W2
// fused over HID chunks so the 16384x4096 hidden never leaves LDS.
//   W1T: (HIDD x HALFD) bf16 row-major  == W1^T
//   W2T: (HALFD x HIDD) bf16 row-major  == W2^T
// ---------------------------------------------------------------------------
static __device__ __forceinline__ void run_branch(
    const __bf16* __restrict__ W1T, const float* __restrict__ b1,
    const __bf16* __restrict__ W2T,
    const __bf16* __restrict__ aT,   // LDS: MT x LDA bf16
    __bf16* __restrict__ hT,         // LDS: MT x LDH bf16
    v8f* __restrict__ acc,           // 8 accumulator tiles
    int wave, int lane) {
  const int half = lane >> 4;        // K-half selector per WMMA operand layout
  const int ln   = lane & 15;
  const int mi   = wave >> 2;        // row group (0..1): rows mi*16..mi*16+15
  const int wc   = wave & 3;         // col group (0..3)
  const int arow = mi * 16 + ln;     // A-operand row held by this lane

  for (int kc = 0; kc < HIDD; kc += KC) {
    // ---- GEMM1: h(MT x KC) = relu(A @ W1[:, kc:kc+KC] + b1) ----
    v8f hacc0 = vzero8();
    v8f hacc1 = vzero8();
    const int n0 = wc * 32;          // this wave's 2 local col tiles: n0, n0+16
    for (int ks = 0; ks < HALFD; ks += 32) {
      const __bf16* ap = aT + arow * LDA + ks + half * 8;
      v16bf af = load_frag(ap, ap + 16);
      const __bf16* bp0 = W1T + (size_t)(kc + n0 + ln) * HALFD + ks + half * 8;
      v16bf bf0 = load_frag(bp0, bp0 + 16);
      hacc0 = __builtin_amdgcn_wmma_f32_16x16x32_bf16(
                  false, af, false, bf0, (short)0, hacc0, false, false);
      const __bf16* bp1 = W1T + (size_t)(kc + n0 + 16 + ln) * HALFD + ks + half * 8;
      v16bf bf1 = load_frag(bp1, bp1 + 16);
      hacc1 = __builtin_amdgcn_wmma_f32_16x16x32_bf16(
                  false, af, false, bf1, (short)0, hacc1, false, false);
    }
    // bias + ReLU + stage bf16 hidden chunk to LDS
    {
      const float bias0 = b1[kc + n0 + ln];
      const float bias1 = b1[kc + n0 + 16 + ln];
      #pragma unroll
      for (int r = 0; r < 8; ++r) {
        const int m = mi * 16 + half * 8 + r;     // C-layout row mapping
        float v0 = hacc0[r] + bias0; v0 = v0 > 0.f ? v0 : 0.f;
        float v1 = hacc1[r] + bias1; v1 = v1 > 0.f ? v1 : 0.f;
        hT[m * LDH + n0 + ln]      = (__bf16)v0;
        hT[m * LDH + n0 + 16 + ln] = (__bf16)v1;
      }
    }
    __syncthreads();
    // ---- GEMM2: acc += h(16 x KC) @ W2[kc:kc+KC, wc*128 : wc*128+128] ----
    for (int ks = 0; ks < KC; ks += 32) {
      const __bf16* hp = hT + arow * LDH + ks + half * 8;
      v16bf hf = load_frag(hp, hp + 16);
      #pragma unroll
      for (int nt = 0; nt < 8; ++nt) {
        const int n = wc * 128 + nt * 16 + ln;
        const __bf16* bp = W2T + (size_t)n * HIDD + kc + ks + half * 8;
        v16bf bf = load_frag(bp, bp + 16);
        acc[nt] = __builtin_amdgcn_wmma_f32_16x16x32_bf16(
                      false, hf, false, bf, (short)0, acc[nt], false, false);
      }
    }
    __syncthreads();   // protect hT before next chunk overwrites it
  }
}

// ---------------------------------------------------------------------------
// Fused coupling kernel: both branches in registers; epilogue writes
// interleaved z_out (float2) and logdet via LDS atomics.
// ---------------------------------------------------------------------------
__global__ __launch_bounds__(NTHREADS)
void coupling_fused_kernel(const float* __restrict__ z,
                           const __bf16* __restrict__ W1sT, const float* __restrict__ b1s,
                           const __bf16* __restrict__ W2sT, const float* __restrict__ b2s,
                           const __bf16* __restrict__ W1tT, const float* __restrict__ b1t,
                           const __bf16* __restrict__ W2tT, const float* __restrict__ b2t,
                           float* __restrict__ zout, float* __restrict__ logdet) {
  extern __shared__ char smem[];
  __bf16* aT    = (__bf16*)smem;                                      // MT*LDA
  __bf16* hT    = (__bf16*)(smem + (size_t)MT * LDA * sizeof(__bf16));// MT*LDH
  float*  ldacc = (float*)(smem + (size_t)MT * LDA * sizeof(__bf16)
                                + (size_t)MT * LDH * sizeof(__bf16)); // MT

  const int tid  = threadIdx.x;
  const int wave = tid >> 5;
  const int lane = tid & 31;
  const int r0   = blockIdx.x * MT;

  // phase 0: stage a = z[:, 0::2] (bf16) into LDS; zero logdet accumulator
  const float2* z2 = (const float2*)z;
  for (int idx = tid; idx < MT * HALFD; idx += NTHREADS) {
    const int m = idx >> 9;            // idx / 512
    const int k = idx & (HALFD - 1);
    const float2 ab = z2[(size_t)(r0 + m) * HALFD + k];
    aT[m * LDA + k] = (__bf16)ab.x;
  }
  if (tid < MT) ldacc[tid] = 0.f;
  __syncthreads();

  v8f acc_t[8], acc_s[8];
  #pragma unroll
  for (int i = 0; i < 8; ++i) { acc_t[i] = vzero8(); acc_s[i] = vzero8(); }

  run_branch(W1tT, b1t, W2tT, aT, hT, acc_t, wave, lane);
  run_branch(W1sT, b1s, W2sT, aT, hT, acc_s, wave, lane);

  // epilogue: b_out = b*exp(s)+t, interleave with a, accumulate logdet
  const int half = lane >> 4;
  const int ln   = lane & 15;
  const int mi   = wave >> 2;
  const int wc   = wave & 3;
  float rowsum[8];
  #pragma unroll
  for (int r = 0; r < 8; ++r) rowsum[r] = 0.f;

  float2* zout2 = (float2*)zout;
  #pragma unroll
  for (int nt = 0; nt < 8; ++nt) {
    const int n  = wc * 128 + nt * 16 + ln;
    const float bs = b2s[n];
    const float bt = b2t[n];
    #pragma unroll
    for (int r = 0; r < 8; ++r) {
      const int m   = mi * 16 + half * 8 + r;   // C-layout row mapping
      const int row = r0 + m;
      const float sval = acc_s[nt][r] + bs;
      const float tval = acc_t[nt][r] + bt;
      rowsum[r] += sval;
      const float2 ab = z2[(size_t)row * HALFD + n];
      float2 o;
      o.x = ab.x;                                // a passes through exactly
      o.y = ab.y * __expf(sval) + tval;          // b_out
      zout2[(size_t)row * HALFD + n] = o;
    }
  }
  #pragma unroll
  for (int r = 0; r < 8; ++r)
    atomicAdd(&ldacc[mi * 16 + half * 8 + r], rowsum[r]);   // ds_add_f32
  __syncthreads();
  if (tid < MT) logdet[r0 + tid] = ldacc[tid];
}

// ---------------------------------------------------------------------------
extern "C" void kernel_launch(void* const* d_in, const int* in_sizes, int n_in,
                              void* d_out, int out_size, void* d_ws, size_t ws_size,
                              hipStream_t stream) {
  const float* z   = (const float*)d_in[0];
  const float* W1s = (const float*)d_in[1];
  const float* b1s = (const float*)d_in[2];
  const float* W2s = (const float*)d_in[3];
  const float* b2s = (const float*)d_in[4];
  const float* W1t = (const float*)d_in[5];
  const float* b1t = (const float*)d_in[6];
  const float* W2t = (const float*)d_in[7];
  const float* b2t = (const float*)d_in[8];

  float* zout   = (float*)d_out;
  float* logdet = zout + (size_t)BROWS * DCOL;

  // workspace: 4 transposed bf16 weight matrices, 2M elems (4 MB) each
  __bf16* ws   = (__bf16*)d_ws;
  const size_t WELEMS = (size_t)HALFD * HIDD;   // 2 * 1024 * 1024
  __bf16* W1sT = ws;                 // (HIDD x HALFD)
  __bf16* W2sT = ws + 1 * WELEMS;    // (HALFD x HIDD)
  __bf16* W1tT = ws + 2 * WELEMS;
  __bf16* W2tT = ws + 3 * WELEMS;

  const dim3 tpb(32, 8);
  transpose_cvt_kernel<<<dim3(HIDD / 32, HALFD / 32), tpb, 0, stream>>>(W1s, W1sT, HALFD, HIDD);
  transpose_cvt_kernel<<<dim3(HALFD / 32, HIDD / 32), tpb, 0, stream>>>(W2s, W2sT, HIDD, HALFD);
  transpose_cvt_kernel<<<dim3(HIDD / 32, HALFD / 32), tpb, 0, stream>>>(W1t, W1tT, HALFD, HIDD);
  transpose_cvt_kernel<<<dim3(HALFD / 32, HIDD / 32), tpb, 0, stream>>>(W2t, W2tT, HIDD, HALFD);

  const size_t lds_bytes = (size_t)MT * LDA * sizeof(__bf16)
                         + (size_t)MT * LDH * sizeof(__bf16)
                         + (size_t)MT * sizeof(float);        // 42112 B
  coupling_fused_kernel<<<BROWS / MT, NTHREADS, lds_bytes, stream>>>(
      z, W1sT, b1s, W2sT, b2s, W1tT, b1t, W2tT, b2t, zout, logdet);
}
